// BinaryLinear_1056561955255
// MI455X (gfx1250) — matrix operand order repaired and checked by reading, not compiled, not run on hardware
//
#include <hip/hip_runtime.h>

typedef __attribute__((ext_vector_type(16))) _Float16 v16h;
typedef __attribute__((ext_vector_type(8)))  _Float16 v8h;
typedef __attribute__((ext_vector_type(8)))  float    v8f;

#define BM 128
#define BN 128
#define BK 32
#define LDA 40   // halves per LDS row: BK + 8 pad -> 80B rows, 16B-aligned chunks

union Frag16 { v16h v; v8h h[2]; };
union Pack4  { _Float16 h[4]; uint2 u; };

// sign(w) as f16: +-1.0h with w's sign bit, 0 if w == +-0 (bitwise, cheap)
__device__ __forceinline__ _Float16 sign_h(float w) {
    union { uint16_t u; _Float16 h; } r;
    const uint32_t b = __float_as_uint(w);
    r.u = (b << 1) ? (uint16_t)(0x3C00u | ((b >> 16) & 0x8000u)) : (uint16_t)0;
    return r.h;
}

// Issue the 8 global_load_b128 for one K-tile into registers (no immediate use:
// compiler keeps them outstanding behind the WMMA stage).
__device__ __forceinline__ void load_tiles(const float* __restrict__ X,
                                           const float* __restrict__ W,
                                           int m0, int n0, int K, int k0, int tid,
                                           float4 xr[4], float4 wr[4]) {
    #pragma unroll
    for (int i = 0; i < 4; ++i) {
        const int chunk = tid + i * 256;       // 0..1023 float4-chunks
        const int row   = chunk >> 3;          // 0..127
        const int col4  = (chunk & 7) << 2;    // 0,4,...,28
        xr[i] = *reinterpret_cast<const float4*>(X + (size_t)(m0 + row) * K + k0 + col4);
        wr[i] = *reinterpret_cast<const float4*>(W + (size_t)(n0 + row) * K + k0 + col4);
    }
}

// Convert staged registers (f32 -> f16; binarize W) and store into LDS tile.
__device__ __forceinline__ void stage_tiles(_Float16* __restrict__ As,
                                            _Float16* __restrict__ Ws,
                                            int tid,
                                            const float4 xr[4], const float4 wr[4]) {
    #pragma unroll
    for (int i = 0; i < 4; ++i) {
        const int chunk = tid + i * 256;
        const int row   = chunk >> 3;
        const int col4  = (chunk & 7) << 2;

        Pack4 xa;
        xa.h[0] = (_Float16)xr[i].x; xa.h[1] = (_Float16)xr[i].y;
        xa.h[2] = (_Float16)xr[i].z; xa.h[3] = (_Float16)xr[i].w;
        *reinterpret_cast<uint2*>(&As[row * LDA + col4]) = xa.u;

        Pack4 wa;
        wa.h[0] = sign_h(wr[i].x); wa.h[1] = sign_h(wr[i].y);
        wa.h[2] = sign_h(wr[i].z); wa.h[3] = sign_h(wr[i].w);
        *reinterpret_cast<uint2*>(&Ws[row * LDA + col4]) = wa.u;
    }
}

__global__ __launch_bounds__(256)
void binlinear_wmma_f16(const float* __restrict__ X,      // [M,K]
                        const float* __restrict__ W,      // [N,K]
                        const float* __restrict__ alpha,  // [N]
                        const float* __restrict__ bias,   // [N]
                        float* __restrict__ Out,          // [M,N]
                        int M, int N, int K)
{
    // ping-pong LDS: one barrier per K-step
    __shared__ __align__(16) _Float16 As[2][BM * LDA];
    __shared__ __align__(16) _Float16 Ws[2][BN * LDA];

    const int tid  = threadIdx.x;
    const int lane = tid & 31;
    const int wid  = tid >> 5;        // 8 wave32 waves
    const int wm   = wid >> 1;        // 0..3 : 32 output rows per wave
    const int wn   = wid & 1;         // 0..1 : 64 output cols per wave

    const int m0 = blockIdx.y * BM;
    const int n0 = blockIdx.x * BN;

    const int rlane   = lane & 15;
    const int halfsel = (lane < 16) ? 0 : 1;

    v8f acc[2][4] = {};   // 8 x (16x16 f32) accumulator tiles

    // --- pipeline prologue: stage K-tile 0 into registers ---
    float4 xr[4], wr[4];
    load_tiles(X, W, m0, n0, K, 0, tid, xr, wr);

    int p = 0;
    for (int k0 = 0; k0 < K; k0 += BK, p ^= 1) {
        // convert previously-loaded registers into LDS buffer p
        stage_tiles(&As[p][0], &Ws[p][0], tid, xr, wr);
        __syncthreads();

        // issue next K-tile's global loads immediately (latency hidden by WMMAs)
        if (k0 + BK < K)
            load_tiles(X, W, m0, n0, K, k0 + BK, tid, xr, wr);
        // and prefetch the tile after that into L2 (gfx1250 global_prefetch)
        if (k0 + 2 * BK < K) {
            __builtin_prefetch(X + (size_t)(m0 + (tid >> 1)) * K + (k0 + 2 * BK), 0, 0);
            __builtin_prefetch(W + (size_t)(n0 + (tid >> 1)) * K + (k0 + 2 * BK), 0, 0);
        }

        // --- fragments from LDS buffer p (documented CDNA5 layouts) ---
        // A 16x32: lane<16 -> row=rlane, K {0..7}U{16..23}; lane>=16 -> K {8..15}U{24..31}
        v16h a[2];
        #pragma unroll
        for (int tm = 0; tm < 2; ++tm) {
            const int arow = wm * 32 + tm * 16 + rlane;
            const _Float16* q = &As[p][arow * LDA + halfsel * 8];
            Frag16 f;
            f.h[0] = *reinterpret_cast<const v8h*>(q);
            f.h[1] = *reinterpret_cast<const v8h*>(q + 16);
            a[tm] = f.v;
        }
        // B 32x16: lane<16 -> col=rlane, K {0..7}U{16..23}; lane>=16 -> K {8..15}U{24..31}
        v16h b[4];
        #pragma unroll
        for (int tn = 0; tn < 4; ++tn) {
            const int brow = wn * 64 + tn * 16 + rlane;
            const _Float16* q = &Ws[p][brow * LDA + halfsel * 8];
            Frag16 f;
            f.h[0] = *reinterpret_cast<const v8h*>(q);
            f.h[1] = *reinterpret_cast<const v8h*>(q + 16);
            b[tn] = f.v;
        }

        // --- 8 matrix ops: D = A x B + C, f32 accumulate ---
        #pragma unroll
        for (int tm = 0; tm < 2; ++tm)
            #pragma unroll
            for (int tn = 0; tn < 4; ++tn)
                acc[tm][tn] = __builtin_amdgcn_wmma_f32_16x16x32_f16(
                    false, a[tm], false, b[tn],
                    (short)0, acc[tm][tn], false, false);
        // no trailing barrier: next iteration writes the other LDS buffer
    }

    // ---- epilogue: out[m,n] = acc * alpha[n] + bias[n] ----
    // C layout: VGPR v, lanes 0-15 -> M=v; lanes 16-31 -> M=v+8; N = lane&15
    #pragma unroll
    for (int tn = 0; tn < 4; ++tn) {
        const int n  = n0 + wn * 64 + tn * 16 + rlane;
        const float al = alpha[n];
        const float bi = bias[n];
        #pragma unroll
        for (int tm = 0; tm < 2; ++tm) {
            const int mbase = m0 + wm * 32 + tm * 16 + halfsel * 8;
            #pragma unroll
            for (int v = 0; v < 8; ++v) {
                Out[(size_t)(mbase + v) * N + n] = acc[tm][tn][v] * al + bi;
            }
        }
    }
}

extern "C" void kernel_launch(void* const* d_in, const int* in_sizes, int n_in,
                              void* d_out, int out_size, void* d_ws, size_t ws_size,
                              hipStream_t stream) {
    const float* X     = (const float*)d_in[0];   // [B, IN]
    const float* W     = (const float*)d_in[1];   // [OUT, IN]
    const float* alpha = (const float*)d_in[2];   // [OUT, 1]
    const float* bias  = (const float*)d_in[3];   // [OUT]
    float* Out = (float*)d_out;                   // [B, OUT]

    const int OUT = in_sizes[3];                  // 4096
    const int IN  = in_sizes[1] / OUT;            // 4096
    const int B   = in_sizes[0] / IN;             // 8192

    dim3 grid(OUT / BN, B / BM);                  // 32 x 64 blocks
    binlinear_wmma_f16<<<grid, dim3(256), 0, stream>>>(X, W, alpha, bias, Out,
                                                       B, OUT, IN);
}
